// PairEmbed_64347200028851
// MI455X (gfx1250) — matrix operand order repaired and checked by reading, not compiled, not run on hardware
//
#include <hip/hip_runtime.h>

#define NUM_ELEMENTS_C 100
#define EMBED_DIM_C    256
#define HIDDEN_DIM_C   512
#define NUM_GAUSS_C    50
#define NUM_HEADS_C    8
#define RBF_RADIUS_C   12.0f
#define KDIM_C         (HIDDEN_DIM_C + EMBED_DIM_C)   /* 768 */

#define KT_RBF 2                      /* K=50 padded to 64 -> 2 k-tiles of 32 */
#define KT_W1  (KDIM_C / 32)          /* 24 */
#define KT_W2  (HIDDEN_DIM_C / 32)    /* 16 */
#define NT_H   (HIDDEN_DIM_C / 16)    /* 32 */

typedef _Float16 v8h  __attribute__((ext_vector_type(8)));
typedef _Float16 v16h __attribute__((ext_vector_type(16)));
typedef float    v8f  __attribute__((ext_vector_type(8)));

/* packed B-fragment pools, sizes in halves */
#define PK_RBF_HALVES (KT_RBF * NT_H * 512)   /* 32768  */
#define PK_W1_HALVES  (KT_W1  * NT_H * 512)   /* 393216 */
#define PK_W2_HALVES  (KT_W2  * 1    * 512)   /* 8192   */

/* ---------------------------------------------------------------------------
 * Pack a row-major f32 matrix W[Ksrc][Nsrc] into f16 WMMA B-fragments.
 * Fragment (kt, nt): 32 lanes x 16 halves contiguous per lane.
 * Lane layout (V_WMMA_*_16X16X32 B, 32x16): n = nt*16 + (lane&15);
 *   lanes 0-15 hold K = kt*32 + 0..7 then 16..23, lanes 16-31 hold K = kt*32 + 8..15 then 24..31,
 *   halves within a lane ordered by K.
 * Out-of-range (K >= Ksrc or N >= Nsrc) elements are zero padding.
 * ------------------------------------------------------------------------- */
__global__ void pack_b_kernel(const float* __restrict__ src,
                              _Float16* __restrict__ dst,
                              int Ksrc, int Nsrc, int NT, int total) {
  int i = blockIdx.x * 256 + threadIdx.x;
  if (i >= total) return;
  int frag = i >> 9;          /* /512 halves per fragment */
  int r    = i & 511;
  int lane = r >> 4;
  int tt   = r & 15;
  int nt = frag % NT;
  int kt = frag / NT;
  int hh = lane >> 4;
  int n0 = lane & 15;
  int k = kt * 32 + ((tt < 8) ? (8 * hh + tt) : (16 + 8 * hh + (tt - 8)));
  int n = nt * 16 + n0;
  float v = (k < Ksrc && n < Nsrc) ? src[(size_t)k * Nsrc + n] : 0.0f;
  dst[i] = (_Float16)v;
}

__device__ __forceinline__ v16h hcat(v8h lo, v8h hi) {
  return __builtin_shufflevector(lo, hi, 0,1,2,3,4,5,6,7,8,9,10,11,12,13,14,15);
}

/* ---------------------------------------------------------------------------
 * Main fused kernel: 256 threads = 8 waves, 64 edges per block.
 *   phase 1: emb gather -> LDS x[:,512:768]; rbf = g @ W_rbf + b -> LDS x[:,0:512]
 *   phase 2: h = silu(x @ W1 + b1) * rbf  -> LDS h[64][512] (f16)
 *   phase 3: att = h @ W2 + b2 -> out[head*E + e]
 * LDS: x 64x768 f16 (96KB) + h 64x512 f16 (64KB) = 160KB (2 blocks / WGP).
 * ------------------------------------------------------------------------- */
__global__ __launch_bounds__(256)
void pairembed_main(const int* __restrict__ anum,
                    const int* __restrict__ rowi,
                    const int* __restrict__ coli,
                    const float* __restrict__ dist,
                    const _Float16* __restrict__ wrbf_pk,
                    const float* __restrict__ b_rbf,
                    const _Float16* __restrict__ w1_pk,
                    const float* __restrict__ b1,
                    const _Float16* __restrict__ w2_pk,
                    const float* __restrict__ b2,
                    const float* __restrict__ emb_table,
                    float* __restrict__ out,
                    int E) {
  extern __shared__ __align__(16) char smem_raw[];
  _Float16* xs = (_Float16*)smem_raw;                        /* [64][768] */
  _Float16* hs = (_Float16*)(smem_raw + 64 * KDIM_C * 2);    /* [64][512] */

  const int t    = threadIdx.x;
  const int wv   = t >> 5;          /* wave id 0..7 */
  const int lane = t & 31;
  const int hh   = lane >> 4;       /* lane half */
  const int m    = lane & 15;       /* A row / B-C column within tile */
  const int e0   = blockIdx.x * 64;

  const float delta = RBF_RADIUS_C / (float)(NUM_GAUSS_C - 1);
  const float coeff = -0.5f / (delta * delta);

  /* ---- phase 1a: pair-embedding gather, f32 -> f16 into xs[:,512:768] ---- */
  {
    int rowl = t >> 2;                          /* 4 threads per edge row */
    int e = e0 + rowl; if (e >= E) e = E - 1;
    int pid = anum[rowi[e]] + NUM_ELEMENTS_C * anum[coli[e]];
    const float2* src = (const float2*)emb_table + (size_t)pid * (EMBED_DIM_C / 2);
    _Float16* dstrow = xs + rowl * KDIM_C + HIDDEN_DIM_C;
    #pragma unroll
    for (int cc = 0; cc < EMBED_DIM_C / 8; ++cc) {
      int c = (t & 3) + 4 * cc;                 /* 0..127 half2 per row */
      float2 v = src[c];
      dstrow[2 * c]     = (_Float16)v.x;
      dstrow[2 * c + 1] = (_Float16)v.y;
    }
  }

  /* ---- phase 1b: rbf = g @ W_rbf + b_rbf via WMMA (wave owns 4 N-tiles) -- */
  {
    /* Build Gaussian A-fragments (16x32 f16 layout) for all 4 M-tiles.
       Lanes 0-15: v0-3 = K 0..7, v4-7 = K 16..23; lanes 16-31: K 8..15 / 24..31 */
    v16h ag[4][KT_RBF];
    #pragma unroll
    for (int mt = 0; mt < 4; ++mt) {
      int e = e0 + mt * 16 + m; if (e >= E) e = E - 1;
      float d = dist[e];
      #pragma unroll
      for (int kt = 0; kt < KT_RBF; ++kt) {
        #pragma unroll
        for (int tt = 0; tt < 16; ++tt) {
          int jv = tt >> 1, odd = tt & 1;
          int kk = kt * 32 + ((jv < 4) ? (8 * hh + 2 * jv + odd)
                                       : (16 + 8 * hh + 2 * (jv - 4) + odd));
          float g = 0.0f;
          if (kk < NUM_GAUSS_C) {
            float u = d - (float)kk * delta;
            g = __expf(coeff * u * u);
          }
          ag[mt][kt][tt] = (_Float16)g;
        }
      }
    }
    #pragma unroll
    for (int ntl = 0; ntl < 4; ++ntl) {
      int nt = wv * 4 + ntl;
      float bb = b_rbf[nt * 16 + m];
      #pragma unroll
      for (int mt = 0; mt < 4; ++mt) {
        v8f c;
        #pragma unroll
        for (int r = 0; r < 8; ++r) c[r] = bb;
        #pragma unroll
        for (int kt = 0; kt < KT_RBF; ++kt) {
          v16h b = *(const v16h*)(wrbf_pk + ((size_t)(kt * NT_H + nt) * 32 + lane) * 16);
          c = __builtin_amdgcn_wmma_f32_16x16x32_f16(false, ag[mt][kt], false, b,
                                                     (short)0, c, false, false);
        }
        /* C layout: VGPR r -> row r (lanes 0-15) / r+8 (lanes 16-31) */
        #pragma unroll
        for (int r = 0; r < 8; ++r) {
          int row = mt * 16 + r + 8 * hh;
          xs[row * KDIM_C + nt * 16 + m] = (_Float16)c[r];
        }
      }
    }
  }

  __syncthreads();

  /* ---- phase 2: h = silu(x @ W1 + b1) * rbf (wave owns 4 N-tiles) ------- */
  {
    v8f acc[4][4];  /* [ntl][mt] */
    #pragma unroll
    for (int ntl = 0; ntl < 4; ++ntl) {
      float bb = b1[(wv * 4 + ntl) * 16 + m];
      #pragma unroll
      for (int mt = 0; mt < 4; ++mt)
        #pragma unroll
        for (int r = 0; r < 8; ++r) acc[ntl][mt][r] = bb;
    }
    for (int kt = 0; kt < KT_W1; ++kt) {
      v16h A[4];
      #pragma unroll
      for (int mt = 0; mt < 4; ++mt) {
        const _Float16* p = xs + (mt * 16 + m) * KDIM_C + kt * 32 + 8 * hh;
        v8h lo = *(const v8h*)p;          /* K 8h .. 8h+7      */
        v8h hi = *(const v8h*)(p + 16);   /* K 16+8h .. 16+8h+7 */
        A[mt] = hcat(lo, hi);
      }
      if (kt + 1 < KT_W1)  /* stream next W1 k-slice through L2 */
        __builtin_prefetch(w1_pk + ((size_t)((kt + 1) * NT_H + wv * 4) * 32 + lane) * 16, 0, 1);
      #pragma unroll
      for (int ntl = 0; ntl < 4; ++ntl) {
        int nt = wv * 4 + ntl;
        v16h B = *(const v16h*)(w1_pk + ((size_t)(kt * NT_H + nt) * 32 + lane) * 16);
        #pragma unroll
        for (int mt = 0; mt < 4; ++mt)
          acc[ntl][mt] = __builtin_amdgcn_wmma_f32_16x16x32_f16(false, A[mt], false, B,
                                                                (short)0, acc[ntl][mt],
                                                                false, false);
      }
    }
    /* epilogue: silu + rbf gate, write f16 h tiles to LDS */
    #pragma unroll
    for (int ntl = 0; ntl < 4; ++ntl) {
      int nt = wv * 4 + ntl;
      #pragma unroll
      for (int mt = 0; mt < 4; ++mt) {
        #pragma unroll
        for (int r = 0; r < 8; ++r) {
          int row = mt * 16 + r + 8 * hh;
          float v = acc[ntl][mt][r];
          float s = v / (1.0f + __expf(-v));           /* silu */
          float gate = (float)xs[row * KDIM_C + nt * 16 + m];
          hs[row * HIDDEN_DIM_C + nt * 16 + m] = (_Float16)(s * gate);
        }
      }
    }
  }

  __syncthreads();

  /* ---- phase 3: att_bias = h @ W2 + b2 (waves 0-3, one M-tile each) ----- */
  if (wv < 4) {
    const int mt = wv;
    v8f c;
    float bb = (m < NUM_HEADS_C) ? b2[m] : 0.0f;
    #pragma unroll
    for (int r = 0; r < 8; ++r) c[r] = bb;
    for (int kt = 0; kt < KT_W2; ++kt) {
      const _Float16* p = hs + (mt * 16 + m) * HIDDEN_DIM_C + kt * 32 + 8 * hh;
      v8h lo = *(const v8h*)p;
      v8h hi = *(const v8h*)(p + 16);
      v16h A = hcat(lo, hi);
      v16h B = *(const v16h*)(w2_pk + ((size_t)kt * 32 + lane) * 16);
      c = __builtin_amdgcn_wmma_f32_16x16x32_f16(false, A, false, B,
                                                 (short)0, c, false, false);
    }
    if (m < NUM_HEADS_C) {
      #pragma unroll
      for (int r = 0; r < 8; ++r) {
        int e = e0 + mt * 16 + r + 8 * hh;
        if (e < E) out[(size_t)m * E + e] = c[r];   /* [masks=1, heads, E] */
      }
    }
  }
}

extern "C" void kernel_launch(void* const* d_in, const int* in_sizes, int n_in,
                              void* d_out, int out_size, void* d_ws, size_t ws_size,
                              hipStream_t stream) {
  const int*   anum      = (const int*)d_in[0];
  const int*   rowi      = (const int*)d_in[1];
  const int*   coli      = (const int*)d_in[2];
  const float* dist      = (const float*)d_in[3];
  const float* W_rbf     = (const float*)d_in[4];
  const float* b_rbf     = (const float*)d_in[5];
  const float* emb_table = (const float*)d_in[6];
  const float* W1        = (const float*)d_in[7];
  const float* b1        = (const float*)d_in[8];
  const float* W2        = (const float*)d_in[9];
  const float* b2        = (const float*)d_in[10];
  float* out = (float*)d_out;
  const int E = in_sizes[3];

  _Float16* wrbf_pk = (_Float16*)d_ws;
  _Float16* w1_pk   = wrbf_pk + PK_RBF_HALVES;
  _Float16* w2_pk   = w1_pk   + PK_W1_HALVES;   /* ~868 KB total in d_ws */

  pack_b_kernel<<<(PK_RBF_HALVES + 255) / 256, 256, 0, stream>>>(
      W_rbf, wrbf_pk, NUM_GAUSS_C, HIDDEN_DIM_C, NT_H, PK_RBF_HALVES);
  pack_b_kernel<<<(PK_W1_HALVES + 255) / 256, 256, 0, stream>>>(
      W1, w1_pk, KDIM_C, HIDDEN_DIM_C, NT_H, PK_W1_HALVES);
  pack_b_kernel<<<(PK_W2_HALVES + 255) / 256, 256, 0, stream>>>(
      W2, w2_pk, HIDDEN_DIM_C, NUM_HEADS_C, 1, PK_W2_HALVES);

  size_t shmem = (size_t)64 * KDIM_C * 2 + (size_t)64 * HIDDEN_DIM_C * 2;  /* 160 KB */
  hipFuncSetAttribute((const void*)pairembed_main,
                      hipFuncAttributeMaxDynamicSharedMemorySize, (int)shmem);

  int blocks = (E + 63) / 64;
  pairembed_main<<<blocks, 256, shmem, stream>>>(
      anum, rowi, coli, dist, wrbf_pk, b_rbf, w1_pk, b1, w2_pk, b2,
      emb_table, out, E);

  (void)n_in; (void)out_size; (void)ws_size;
}
